// RobiiNet_22840636080371
// MI455X (gfx1250) — matrix operand order repaired and checked by reading, not compiled, not run on hardware
//
#include <hip/hip_runtime.h>
#include <hip/hip_bf16.h>
#include <math.h>

// ---- problem constants (match reference) ----
#define BATCH   16
#define NVIS    4096
#define NPIX2   16384
#define DEPTH_CT 10
#define ALPHA_CT 0.001f

typedef __attribute__((ext_vector_type(16))) _Float16 v16h;
typedef __attribute__((ext_vector_type(8)))  float    v8f;

// ---------------- WMMA helpers (gfx1250, wave32) ----------------

static __device__ __forceinline__ v8f wmma_f16(v16h a, v16h b, v8f c) {
  // (neg_a, A, neg_b, B, c_mod, C, reuse_a, reuse_b)
  return __builtin_amdgcn_wmma_f32_16x16x32_f16(false, a, false, b, (short)0, c,
                                                false, false);
}

// A fragment, 16x32 f16, from f32 row-major A (lda elements between rows).
// lanes 0-15: row M=lane, K = kb..kb+15
// lanes 16-31: row M=lane-16, K = kb+16..kb+31
static __device__ __forceinline__ v16h frag_a(const float* __restrict__ A, int lda,
                                              int lane, int kb) {
  const int m  = lane & 15;
  const int ks = (lane >= 16) ? 16 : 0;
  const float* p = A + (size_t)m * lda + kb + ks;
  v16h f;
#pragma unroll
  for (int j = 0; j < 8; ++j) f[j]     = (_Float16)p[j];
#pragma unroll
  for (int j = 0; j < 8; ++j) f[8 + j] = (_Float16)p[8 + j];
  return f;
}

static __device__ __forceinline__ v16h frag_neg(v16h a) {
  v16h f;
#pragma unroll
  for (int j = 0; j < 16; ++j) f[j] = (_Float16)(-a[j]);  // lowers to packed v_xor
  return f;
}

// B fragment for NT GEMM: B[k][n] = W[nrow][k], W row-major with ldw along K.
// lanes 0-15: col n, K = kb..kb+15 ; lanes 16-31: K = kb+16..kb+31
static __device__ __forceinline__ v16h frag_b_nt(const float* __restrict__ W, int ldw,
                                                 int nrow, int lane, int kb) {
  const int ks = (lane >= 16) ? 16 : 0;
  const float* p = W + (size_t)nrow * ldw + kb + ks;
  __builtin_prefetch(p + 32, 0, 1);            // next k-step of this W row (HBM stream)
  v16h f;
#pragma unroll
  for (int j = 0; j < 16; ++j) f[j] = (_Float16)p[j];
  return f;
}

// B fragment for NN GEMM: B[k][p] = W[k][p], W row-major with ldw = row width.
static __device__ __forceinline__ v16h frag_b_nn(const float* __restrict__ W, int ldw,
                                                 int pcol, int lane, int kb) {
  const int ks = (lane >= 16) ? 16 : 0;
  const float* p = W + (size_t)(kb + ks) * ldw + pcol;
  v16h f;
#pragma unroll
  for (int j = 0; j < 16; ++j) f[j] = (_Float16)p[(size_t)j * ldw];
  return f;
}

// ---------------- GEMM kernels ----------------

// zk = xk @ W^T (complex, no conj):  re = xr@Wr^T - xi@Wi^T ; im = xr@Wi^T + xi@Wr^T
// A: [16,K] planes, W: [N,K] planes (row-major, ld=K), C: [16,N] planes.
// Each wave produces TWO adjacent 16-col tiles (A frags amortized over 4 B frags).
__global__ void k_gemm_nt_cplx(const float* __restrict__ Are, const float* __restrict__ Aim,
                               const float* __restrict__ Wre, const float* __restrict__ Wim,
                               float* __restrict__ Cre, float* __restrict__ Cim,
                               int K, int N) {
  const int lane = threadIdx.x & 31;
  const int wtile = blockIdx.x * (blockDim.x >> 5) + (threadIdx.x >> 5);
  const int n0 = wtile * 32;
  if (n0 >= N) return;                       // wave-uniform
  const int nrow0 = n0 + (lane & 15);
  const int nrow1 = nrow0 + 16;
  v8f acc_re0 = {}, acc_im0 = {}, acc_re1 = {}, acc_im1 = {};
  for (int kb = 0; kb < K; kb += 32) {
    v16h ar  = frag_a(Are, K, lane, kb);
    v16h ai  = frag_a(Aim, K, lane, kb);
    v16h ain = frag_neg(ai);
    v16h br0 = frag_b_nt(Wre, K, nrow0, lane, kb);
    v16h bi0 = frag_b_nt(Wim, K, nrow0, lane, kb);
    v16h br1 = frag_b_nt(Wre, K, nrow1, lane, kb);
    v16h bi1 = frag_b_nt(Wim, K, nrow1, lane, kb);
    acc_re0 = wmma_f16(ar,  br0, acc_re0);
    acc_re0 = wmma_f16(ain, bi0, acc_re0);   // - xi@Wi^T
    acc_im0 = wmma_f16(ar,  bi0, acc_im0);
    acc_im0 = wmma_f16(ai,  br0, acc_im0);
    acc_re1 = wmma_f16(ar,  br1, acc_re1);
    acc_re1 = wmma_f16(ain, bi1, acc_re1);
    acc_im1 = wmma_f16(ar,  bi1, acc_im1);
    acc_im1 = wmma_f16(ai,  br1, acc_im1);
  }
  const int mb = (lane >= 16) ? 8 : 0;
  const int nc0 = n0 + (lane & 15);
#pragma unroll
  for (int r = 0; r < 8; ++r) {
    size_t row = (size_t)(mb + r) * N;
    Cre[row + nc0]      = acc_re0[r];
    Cim[row + nc0]      = acc_im0[r];
    Cre[row + nc0 + 16] = acc_re1[r];
    Cim[row + nc0 + 16] = acc_im1[r];
  }
}

// wtemp = sigmoid(xt @ W^T + b)
__global__ void k_gemm_nt_sigmoid(const float* __restrict__ A, const float* __restrict__ W,
                                  const float* __restrict__ bias, float* __restrict__ C,
                                  int K, int N) {
  const int lane = threadIdx.x & 31;
  const int tile = blockIdx.x * (blockDim.x >> 5) + (threadIdx.x >> 5);
  const int n0 = tile * 16;
  if (n0 >= N) return;
  const int nrow = n0 + (lane & 15);
  v8f acc = {};
  for (int kb = 0; kb < K; kb += 32) {
    v16h a = frag_a(A, K, lane, kb);
    v16h b = frag_b_nt(W, K, nrow, lane, kb);
    acc = wmma_f16(a, b, acc);
  }
  const int mb = (lane >= 16) ? 8 : 0;
  const int nc = n0 + (lane & 15);
  const float bv = bias[nc];
#pragma unroll
  for (int r = 0; r < 8; ++r) {
    float v = acc[r] + bv;
    C[(size_t)(mb + r) * N + nc] = 1.0f / (1.0f + __expf(-v));
  }
}

// tau_new = relu(wtemp @ Wu^T + bu + tau_old @ Wm^T + bm)  (two K-loops, one accumulator)
__global__ void k_gemm_nt_dual_relu(const float* __restrict__ A1, const float* __restrict__ W1,
                                    const float* __restrict__ b1,
                                    const float* __restrict__ A2, const float* __restrict__ W2,
                                    const float* __restrict__ b2,
                                    float* __restrict__ C, int K, int N) {
  const int lane = threadIdx.x & 31;
  const int tile = blockIdx.x * (blockDim.x >> 5) + (threadIdx.x >> 5);
  const int n0 = tile * 16;
  if (n0 >= N) return;
  const int nrow = n0 + (lane & 15);
  v8f acc = {};
  for (int kb = 0; kb < K; kb += 32) {
    v16h a = frag_a(A1, K, lane, kb);
    v16h b = frag_b_nt(W1, K, nrow, lane, kb);
    acc = wmma_f16(a, b, acc);
  }
  for (int kb = 0; kb < K; kb += 32) {
    v16h a = frag_a(A2, K, lane, kb);
    v16h b = frag_b_nt(W2, K, nrow, lane, kb);
    acc = wmma_f16(a, b, acc);
  }
  const int mb = (lane >= 16) ? 8 : 0;
  const int nc = n0 + (lane & 15);
  const float bv = b1[nc] + b2[nc];
#pragma unroll
  for (int r = 0; r < 8; ++r) {
    float v = acc[r] + bv;
    C[(size_t)(mb + r) * N + nc] = fmaxf(v, 0.0f);
  }
}

// C (+)= scale * (A @ conj(W)):  re = ar@Wr + ai@Wi ; im = ai@Wr - ar@Wi
// A: [16,K] planes (ld=K), W: [K,N] planes row-major (ld=N), C: [16,N] planes.
// Two adjacent 16-col tiles per wave.
__global__ void k_gemm_nn_conj(const float* __restrict__ Are, const float* __restrict__ Aim,
                               const float* __restrict__ Wre, const float* __restrict__ Wim,
                               float* __restrict__ Cre, float* __restrict__ Cim,
                               int K, int N, float scale, int accumulate) {
  const int lane = threadIdx.x & 31;
  const int wtile = blockIdx.x * (blockDim.x >> 5) + (threadIdx.x >> 5);
  const int p0 = wtile * 32;
  if (p0 >= N) return;
  const int pcol0 = p0 + (lane & 15);
  const int pcol1 = pcol0 + 16;
  v8f acc_re0 = {}, acc_im0 = {}, acc_re1 = {}, acc_im1 = {};
  for (int kb = 0; kb < K; kb += 32) {
    v16h ar  = frag_a(Are, K, lane, kb);
    v16h ai  = frag_a(Aim, K, lane, kb);
    v16h arn = frag_neg(ar);
    v16h br0 = frag_b_nn(Wre, N, pcol0, lane, kb);
    v16h bi0 = frag_b_nn(Wim, N, pcol0, lane, kb);
    v16h br1 = frag_b_nn(Wre, N, pcol1, lane, kb);
    v16h bi1 = frag_b_nn(Wim, N, pcol1, lane, kb);
    acc_re0 = wmma_f16(ar,  br0, acc_re0);
    acc_re0 = wmma_f16(ai,  bi0, acc_re0);   // + ai@Wi  (conj)
    acc_im0 = wmma_f16(ai,  br0, acc_im0);
    acc_im0 = wmma_f16(arn, bi0, acc_im0);   // - ar@Wi  (conj)
    acc_re1 = wmma_f16(ar,  br1, acc_re1);
    acc_re1 = wmma_f16(ai,  bi1, acc_re1);
    acc_im1 = wmma_f16(ai,  br1, acc_im1);
    acc_im1 = wmma_f16(arn, bi1, acc_im1);
  }
  const int mb = (lane >= 16) ? 8 : 0;
  const int pc0 = p0 + (lane & 15);
#pragma unroll
  for (int r = 0; r < 8; ++r) {
    size_t row = (size_t)(mb + r) * N;
    float vr0 = acc_re0[r] * scale, vi0 = acc_im0[r] * scale;
    float vr1 = acc_re1[r] * scale, vi1 = acc_im1[r] * scale;
    if (accumulate) {
      vr0 += Cre[row + pc0];      vi0 += Cim[row + pc0];
      vr1 += Cre[row + pc0 + 16]; vi1 += Cim[row + pc0 + 16];
    }
    Cre[row + pc0]      = vr0;
    Cim[row + pc0]      = vi0;
    Cre[row + pc0 + 16] = vr1;
    Cim[row + pc0 + 16] = vi1;
  }
}

// ---------------- elementwise / reduction kernels ----------------

__global__ void k_zero(float* __restrict__ p, int n) {
  int i = blockIdx.x * blockDim.x + threadIdx.x;
  if (i < n) p[i] = 0.0f;
}

// r = y - zk ; xt = |r|^2
__global__ void k_resid(const float* __restrict__ yr, const float* __restrict__ yi,
                        const float* __restrict__ zr, const float* __restrict__ zi,
                        float* __restrict__ rr, float* __restrict__ ri,
                        float* __restrict__ xt, int n) {
  int i = blockIdx.x * blockDim.x + threadIdx.x;
  if (i >= n) return;
  float a = yr[i] - zr[i];
  float b = yi[i] - zi[i];
  rr[i] = a; ri[i] = b;
  xt[i] = a * a + b * b;
}

// v = tau * r   (complex scaled by real)
__global__ void k_vmul(const float* __restrict__ tau,
                       const float* __restrict__ rr, const float* __restrict__ ri,
                       float* __restrict__ vr, float* __restrict__ vi, int n) {
  int i = blockIdx.x * blockDim.x + threadIdx.x;
  if (i >= n) return;
  float t = tau[i];
  vr[i] = t * rr[i];
  vi[i] = t * ri[i];
}

__global__ void k_sqmax_partial(const float* __restrict__ re, const float* __restrict__ im,
                                float* __restrict__ partial, int n) {
  __shared__ float s[256];
  float m = 0.0f;
  for (int i = blockIdx.x * blockDim.x + threadIdx.x; i < n; i += gridDim.x * blockDim.x) {
    float a = re[i], b = im[i];
    m = fmaxf(m, a * a + b * b);
  }
  s[threadIdx.x] = m;
  __syncthreads();
  for (int off = 128; off > 0; off >>= 1) {
    if ((int)threadIdx.x < off) s[threadIdx.x] = fmaxf(s[threadIdx.x], s[threadIdx.x + off]);
    __syncthreads();
  }
  if (threadIdx.x == 0) partial[blockIdx.x] = s[0];
}

__global__ void k_max_final(const float* __restrict__ partial, int n, float* __restrict__ out) {
  __shared__ float s[256];
  float m = 0.0f;
  for (int i = threadIdx.x; i < n; i += 256) m = fmaxf(m, partial[i]);
  s[threadIdx.x] = m;
  __syncthreads();
  for (int off = 128; off > 0; off >>= 1) {
    if ((int)threadIdx.x < off) s[threadIdx.x] = fmaxf(s[threadIdx.x], s[threadIdx.x + off]);
    __syncthreads();
  }
  if (threadIdx.x == 0) out[0] = sqrtf(s[0]);   // max |x|
}

// xk = sign(x) * relu(|x| - alpha*max|x|)
__global__ void k_softthresh(float* __restrict__ re, float* __restrict__ im,
                             const float* __restrict__ maxval, int n) {
  int i = blockIdx.x * blockDim.x + threadIdx.x;
  if (i >= n) return;
  const float thr = maxval[0] * ALPHA_CT;
  float a = re[i], b = im[i];
  float ax = sqrtf(a * a + b * b);
  float mag = fmaxf(ax - thr, 0.0f);
  float sc = (ax > 0.0f) ? (mag / ax) : 0.0f;
  re[i] = a * sc;
  im[i] = b * sc;
}

__global__ void k_cabs(const float* __restrict__ re, const float* __restrict__ im,
                       float* __restrict__ out, int n) {
  int i = blockIdx.x * blockDim.x + threadIdx.x;
  if (i >= n) return;
  float a = re[i], b = im[i];
  out[i] = sqrtf(a * a + b * b);
}

// ---------------- launch ----------------

extern "C" void kernel_launch(void* const* d_in, const int* in_sizes, int n_in,
                              void* d_out, int out_size, void* d_ws, size_t ws_size,
                              hipStream_t stream) {
  const float* y_re     = (const float*)d_in[0];
  const float* y_im     = (const float*)d_in[1];
  const float* W_re     = (const float*)d_in[2];
  const float* W_im     = (const float*)d_in[3];
  const float* estep_w  = (const float*)d_in[4];
  const float* estep_b  = (const float*)d_in[5];
  const float* update_w = (const float*)d_in[6];
  const float* update_b = (const float*)d_in[7];
  const float* memory_w = (const float*)d_in[8];
  const float* memory_b = (const float*)d_in[9];
  // d_in[10] = depth (device scalar) — fixed DEPTH_CT=10 at compile time
  // (cannot read device memory on host under graph capture).
  (void)in_sizes; (void)n_in; (void)out_size; (void)ws_size;

  const int nbig = BATCH * NPIX2;   // 262144
  const int nsml = BATCH * NVIS;    // 65536

  float* ws = (float*)d_ws;
  float* xk_re = ws;  ws += nbig;
  float* xk_im = ws;  ws += nbig;
  float* zk_re = ws;  ws += nsml;
  float* zk_im = ws;  ws += nsml;
  float* r_re  = ws;  ws += nsml;
  float* r_im  = ws;  ws += nsml;
  float* xt    = ws;  ws += nsml;
  float* wtemp = ws;  ws += nsml;
  float* tau0  = ws;  ws += nsml;
  float* tau1  = ws;  ws += nsml;
  float* v_re  = ws;  ws += nsml;
  float* v_im  = ws;  ws += nsml;
  float* partial = ws; ws += 256;
  float* maxv  = ws;  ws += 1;
  float* tbuf[2] = { tau0, tau1 };

  const dim3 blk(256);                         // 8 wave32 waves per block
  const int gsml  = (nsml + 255) / 256;
  const int gbig  = (nbig + 255) / 256;
  const int gN16  = NVIS  / (16 * 8);          // 32 blocks  (16-wide tiles)
  const int gN32  = NVIS  / (32 * 8);          // 16 blocks  (32-wide tiles, cplx)
  const int gP32  = NPIX2 / (32 * 8);          // 64 blocks  (32-wide tiles, conj)

  // tau = 0
  k_zero<<<gsml, blk, 0, stream>>>(tau0, nsml);
  // x0 = y @ conj(W) / npix2
  k_gemm_nn_conj<<<gP32, blk, 0, stream>>>(y_re, y_im, W_re, W_im, xk_re, xk_im,
                                           NVIS, NPIX2, 1.0f / (float)NPIX2, 0);

  for (int it = 0; it < DEPTH_CT; ++it) {
    float* tin  = tbuf[it & 1];
    float* tout = tbuf[(it + 1) & 1];

    // zk = xk @ W^T        [16,4096] complex
    k_gemm_nt_cplx<<<gN32, blk, 0, stream>>>(xk_re, xk_im, W_re, W_im,
                                             zk_re, zk_im, NPIX2, NVIS);
    // r = y - zk ; xt = |r|^2
    k_resid<<<gsml, blk, 0, stream>>>(y_re, y_im, zk_re, zk_im, r_re, r_im, xt, nsml);
    // wtemp = sigmoid(xt @ estep_w^T + estep_b)
    k_gemm_nt_sigmoid<<<gN16, blk, 0, stream>>>(xt, estep_w, estep_b, wtemp, NVIS, NVIS);
    // tau' = relu(wtemp @ update_w^T + update_b + tau @ memory_w^T + memory_b)
    k_gemm_nt_dual_relu<<<gN16, blk, 0, stream>>>(wtemp, update_w, update_b,
                                                  tin, memory_w, memory_b,
                                                  tout, NVIS, NVIS);
    // v = tau' * r    (algebraic fusion: x = xk + Wt(tau*(y-zk))/nvis)
    k_vmul<<<gsml, blk, 0, stream>>>(tout, r_re, r_im, v_re, v_im, nsml);
    // xk += (v @ conj(W)) / (npix2*nvis)
    k_gemm_nn_conj<<<gP32, blk, 0, stream>>>(v_re, v_im, W_re, W_im, xk_re, xk_im,
                                             NVIS, NPIX2,
                                             1.0f / ((float)NPIX2 * (float)NVIS), 1);
    // global max |x| (two-stage over squared magnitude)
    k_sqmax_partial<<<256, blk, 0, stream>>>(xk_re, xk_im, partial, nbig);
    k_max_final<<<1, blk, 0, stream>>>(partial, 256, maxv);
    // xk = sign(x) * relu(|x| - alpha*max)
    k_softthresh<<<gbig, blk, 0, stream>>>(xk_re, xk_im, maxv, nbig);
  }

  // out = |xk|
  k_cabs<<<gbig, blk, 0, stream>>>(xk_re, xk_im, (float*)d_out, nbig);
}
